// Encoder_68616397521032
// MI455X (gfx1250) — matrix operand order repaired and checked by reading, not compiled, not run on hardware
//
#include <hip/hip_runtime.h>
#include <hip/hip_bf16.h>

// ---------------------------------------------------------------------------
// Transformer encoder for MI455X (gfx1250, wave32, WMMA f32_16x16x32_f16)
// B=128 S=256 D=512 L=4 V=96 H=8 FF=512 DH=64  -> N = B*S = 32768 tokens
// ---------------------------------------------------------------------------

typedef _Float16 h16;
typedef __attribute__((ext_vector_type(16))) _Float16 v16h;
typedef __attribute__((ext_vector_type(8)))  _Float16 v8h;
typedef __attribute__((ext_vector_type(8)))  float    v8f;
typedef __attribute__((ext_vector_type(4)))  float    v4f;
typedef __attribute__((ext_vector_type(4)))  int      v4i;

#define NTOK   32768
#define DM     512
#define SEQ    256
#define NHEAD  8
#define DH     64
#define NLAYER 4
#define BN_EPS 1e-3f

// ---------------------------------------------------------------------------
// Async global->LDS staging (ASYNCcnt path), guarded: falls back to the
// register round-trip if the builtin is not available in this toolchain.
// Signature (from compiler diagnostic): (v4i AS1*, v4i AS3*, imm off, imm cpol)
// ---------------------------------------------------------------------------
#if defined(__has_builtin)
#if __has_builtin(__builtin_amdgcn_global_load_async_to_lds_b128) && \
    __has_builtin(__builtin_amdgcn_s_wait_asynccnt)
#define USE_ASYNC_LDS 1
#endif
#endif
#ifndef USE_ASYNC_LDS
#define USE_ASYNC_LDS 0
#endif

typedef __attribute__((address_space(1))) v4i g_v4i;
typedef __attribute__((address_space(3))) v4i l_v4i;
typedef __attribute__((address_space(1))) const h16 g_h16c;
typedef __attribute__((address_space(3))) h16 l_h16;

// copy 16 bytes (8 halves) global -> LDS
__device__ __forceinline__ void cp16(h16* lds, const h16* g) {
#if USE_ASYNC_LDS
  __builtin_amdgcn_global_load_async_to_lds_b128(
      (g_v4i*)(g_h16c*)g, (l_v4i*)(l_h16*)lds, 0, 0);
#else
  *(v8h*)lds = *(const v8h*)g;
#endif
}

__device__ __forceinline__ void cp_wait() {
#if USE_ASYNC_LDS
  __builtin_amdgcn_s_wait_asynccnt(0);
#endif
}

// ---------------------------------------------------------------------------
// WMMA helpers (layouts per CDNA5 ISA §7.12.2)
// ---------------------------------------------------------------------------
__device__ __forceinline__ v8f wmma32(v16h a, v16h b, v8f c) {
  return __builtin_amdgcn_wmma_f32_16x16x32_f16(false, a, false, b, (short)0, c,
                                                false, false);
}

// A fragment: 16x32 f16 tile, row-major source, stride ld (halves).
// lane m<16 holds row m: h[0..7]=K(kb..kb+7), h[8..15]=K(kb+16..kb+23), kb=0|8.
__device__ __forceinline__ v16h load_a(const h16* __restrict__ p0, int ld) {
  int lane = threadIdx.x & 31;
  int m = lane & 15;
  int kb = (lane & 16) ? 8 : 0;
  const h16* p = p0 + m * ld + kb;
  v8h lo = *(const v8h*)(p);
  v8h hi = *(const v8h*)(p + 16);
  v16h r;
#pragma unroll
  for (int i = 0; i < 8; ++i) { r[i] = lo[i]; r[i + 8] = hi[i]; }
  return r;
}

// B fragment: 32x16 f16 tile, row-major source, stride ld (halves).
// lane n<16 holds column n: h[0..7]=K(kb..kb+7), h[8..15]=K(kb+16..kb+23), kb=0|8.
__device__ __forceinline__ v16h load_b(const h16* __restrict__ p0, int ld) {
  int lane = threadIdx.x & 31;
  int n = lane & 15;
  int kb = (lane & 16) ? 8 : 0;
  const h16* p = p0 + kb * ld + n;
  v16h r;
#pragma unroll
  for (int i = 0; i < 8; ++i) { r[i] = p[i * ld]; r[i + 8] = p[(i + 16) * ld]; }
  return r;
}

__device__ __forceinline__ v8f zero8() {
  v8f z = {0.f, 0.f, 0.f, 0.f, 0.f, 0.f, 0.f, 0.f};
  return z;
}

// ---------------------------------------------------------------------------
// GEMM: C[32768 x 512] = A[32768 x 512](f16) * B[512 x 512](f16) + bias
// mode 0: f16 out   mode 1: relu -> f16 out   mode 2: +resid(f32) -> f32 out
// Block 256 thr (8 waves). Block tile 128x128, wave tile 32x64, BK=64
// (2 WMMA K-substeps per stage), double-buffered LDS, async LDS staging.
// ---------------------------------------------------------------------------
__launch_bounds__(256)
__global__ void gemm_kernel(const h16* __restrict__ A, const h16* __restrict__ Bm,
                            const float* __restrict__ bias,
                            const float* __restrict__ resid,
                            float* __restrict__ outF, h16* __restrict__ outH,
                            int mode) {
  const int K = DM, NC = DM;
  __shared__ h16 As[2][128][72];   // 128 x 64 + 8 pad
  __shared__ h16 Bs[2][64][136];   // 64 x 128 + 8 pad

  int tid = threadIdx.x;
  int wave = tid >> 5, lane = tid & 31;
  int wm = wave >> 1, wn = wave & 1;
  long bm = (long)blockIdx.x * 128;
  int  bn = blockIdx.y * 128;

  int arow = tid >> 1, acol = (tid & 1) * 32;  // 128 rows x 64 halves
  int brow = tid >> 2, bcol = (tid & 3) * 32;  // 64 rows x 128 halves

  v8f acc[2][4];
#pragma unroll
  for (int i = 0; i < 2; ++i)
#pragma unroll
    for (int j = 0; j < 4; ++j) acc[i][j] = zero8();

  auto stage = [&](int buf, int k0) {
    const h16* sa = A + (bm + arow) * K + k0 + acol;
#pragma unroll
    for (int c = 0; c < 4; ++c) cp16(&As[buf][arow][acol + 8 * c], sa + 8 * c);
    const h16* sb = Bm + (long)(k0 + brow) * NC + bn + bcol;
#pragma unroll
    for (int c = 0; c < 4; ++c) cp16(&Bs[buf][brow][bcol + 8 * c], sb + 8 * c);
  };

  stage(0, 0);
  cp_wait();
  __syncthreads();

  for (int kt = 0; kt < 8; ++kt) {
    int cur = kt & 1, nxt = cur ^ 1;
    if (kt < 7) stage(nxt, (kt + 1) * 64);
#pragma unroll
    for (int ks = 0; ks < 2; ++ks) {
      v16h af0 = load_a(&As[cur][wm * 32][ks * 32], 72);
      v16h af1 = load_a(&As[cur][wm * 32 + 16][ks * 32], 72);
#pragma unroll
      for (int nt = 0; nt < 4; ++nt) {
        v16h bf = load_b(&Bs[cur][ks * 32][wn * 64 + nt * 16], 136);
        acc[0][nt] = wmma32(af0, bf, acc[0][nt]);
        acc[1][nt] = wmma32(af1, bf, acc[1][nt]);
      }
    }
    cp_wait();
    __syncthreads();
  }

  int lm = lane & 15;
  int lo8 = (lane & 16) ? 8 : 0;
#pragma unroll
  for (int mt = 0; mt < 2; ++mt)
#pragma unroll
    for (int nt = 0; nt < 4; ++nt)
#pragma unroll
      for (int r = 0; r < 8; ++r) {
        long row = bm + wm * 32 + mt * 16 + r + lo8;
        int  col = bn + wn * 64 + nt * 16 + lm;
        float v = acc[mt][nt][r] + bias[col];
        if (mode == 1) v = fmaxf(v, 0.f);
        long idx = row * NC + col;
        if (mode == 2) outF[idx] = v + resid[idx];
        else           outH[idx] = (h16)v;
      }
}

// ---------------------------------------------------------------------------
// Fused attention. Grid (B*H, 2); block 256 (8 waves, 16 q-rows each).
// K^T (64x256) and V (256x64) staged in LDS; scores + softmax in registers;
// P round-trips through per-wave LDS into A-layout for the P*V GEMM.
// O is written back into the Q buffer (each block reads exactly the region it
// later writes, reads complete before stores issue).
// ---------------------------------------------------------------------------
#define KT_LD 264   // 256 + 8 pad
#define VS_LD 72    // 64 + 8 pad
#define ATTN_LDS_HALVES (64 * KT_LD + 256 * VS_LD + 8 * 16 * KT_LD)

__launch_bounds__(256)
__global__ void attn_kernel(const h16* __restrict__ Q, const h16* __restrict__ Kg,
                            const h16* __restrict__ Vg, h16* __restrict__ O) {
  extern __shared__ char smem_raw[];
  h16* Kt = (h16*)smem_raw;            // [64][KT_LD]  K transposed: [d][s]
  h16* Vs = Kt + 64 * KT_LD;           // [256][VS_LD] V row-major:  [s][d]
  h16* Ps = Vs + 256 * VS_LD;          // [8 waves][16][KT_LD] probabilities

  int tid = threadIdx.x, wave = tid >> 5, lane = tid & 31;
  int bh = blockIdx.x;
  int b = bh / NHEAD, hh = bh % NHEAD;
  long base = (long)b * SEQ * DM + hh * DH;

  // stage K (transposed, via registers) and V (async straight to LDS)
#pragma unroll
  for (int i = 0; i < 4; ++i) {
    int idx = tid + 256 * i;       // 0..1023
    int s = idx >> 2;              // 0..255
    int dc = (idx & 3) * 16;       // 0,16,32,48
    const h16* ks = Kg + base + (long)s * DM + dc;
    v8h a0 = *(const v8h*)ks;
    v8h a1 = *(const v8h*)(ks + 8);
#pragma unroll
    for (int j = 0; j < 8; ++j) {
      Kt[(dc + j) * KT_LD + s]     = a0[j];
      Kt[(dc + 8 + j) * KT_LD + s] = a1[j];
    }
    const h16* vs = Vg + base + (long)s * DM + dc;
    cp16(&Vs[s * VS_LD + dc], vs);
    cp16(&Vs[s * VS_LD + dc + 8], vs + 8);
  }
  cp_wait();
  __syncthreads();

  int q0 = blockIdx.y * 128 + wave * 16;
  const h16* qp = Q + base + (long)q0 * DM;
  v16h aq0 = load_a(qp, DM);        // d = 0..31
  v16h aq1 = load_a(qp + 32, DM);   // d = 32..63

  // scores: 16 q-rows x 256 keys, 16 C tiles
  v8f sc[16];
#pragma unroll
  for (int t = 0; t < 16; ++t) {
    v8f c = zero8();
    c = wmma32(aq0, load_b(&Kt[0 * KT_LD + t * 16], KT_LD), c);
    c = wmma32(aq1, load_b(&Kt[32 * KT_LD + t * 16], KT_LD), c);
    sc[t] = c;
  }

  // softmax over keys (scale 1/sqrt(64) folded into exp argument)
  float mx[8], sm[8];
#pragma unroll
  for (int r = 0; r < 8; ++r) {
    float m = sc[0][r];
#pragma unroll
    for (int t = 1; t < 16; ++t) m = fmaxf(m, sc[t][r]);
    for (int d = 1; d < 16; d <<= 1) m = fmaxf(m, __shfl_xor(m, d, 32));
    mx[r] = m;
  }
#pragma unroll
  for (int r = 0; r < 8; ++r) {
    float s = 0.f;
#pragma unroll
    for (int t = 0; t < 16; ++t) {
      float e = __expf((sc[t][r] - mx[r]) * 0.125f);
      sc[t][r] = e;
      s += e;
    }
    for (int d = 1; d < 16; d <<= 1) s += __shfl_xor(s, d, 32);
    sm[r] = s;
  }

  // P -> per-wave LDS (row-major [16][256]) for A-layout reload
  h16* pw = Ps + wave * (16 * KT_LD);
  int lm = lane & 15, lo8 = (lane & 16) ? 8 : 0;
#pragma unroll
  for (int t = 0; t < 16; ++t)
#pragma unroll
    for (int r = 0; r < 8; ++r)
      pw[(r + lo8) * KT_LD + t * 16 + lm] = (h16)sc[t][r];

  // O = P * V : K-dim = 256 keys in 8 chunks of 32
  v8f oc[4];
#pragma unroll
  for (int nt = 0; nt < 4; ++nt) oc[nt] = zero8();
#pragma unroll
  for (int kt = 0; kt < 8; ++kt) {
    v16h ap = load_a(pw + kt * 32, KT_LD);
#pragma unroll
    for (int nt = 0; nt < 4; ++nt) {
      v16h bv = load_b(&Vs[(kt * 32) * VS_LD + nt * 16], VS_LD);
      oc[nt] = wmma32(ap, bv, oc[nt]);
    }
  }

  float inv[8];
#pragma unroll
  for (int r = 0; r < 8; ++r) inv[r] = 1.f / sm[r];

  h16* op = O + base + (long)q0 * DM;
#pragma unroll
  for (int nt = 0; nt < 4; ++nt)
#pragma unroll
    for (int r = 0; r < 8; ++r)
      op[(long)(r + lo8) * DM + nt * 16 + lm] = (h16)(oc[nt][r] * inv[r]);
}

// ---------------------------------------------------------------------------
// BatchNorm support kernels
// ---------------------------------------------------------------------------
__global__ void bn_stats_kernel(const float* __restrict__ z,
                                float* __restrict__ stats) {
  int f = threadIdx.x;                          // 256 threads -> 2 features
  long r0 = (long)blockIdx.x * 128;             // 256 blocks x 128 rows
  float s0 = 0.f, q0 = 0.f, s1 = 0.f, q1 = 0.f;
  for (int r = 0; r < 128; ++r) {
    const float* row = z + (r0 + r) * DM;
    float a = row[f], b = row[f + 256];
    s0 += a; q0 += a * a;
    s1 += b; q1 += b * b;
  }
  atomicAdd(&stats[f], s0);        atomicAdd(&stats[DM + f], q0);
  atomicAdd(&stats[f + 256], s1);  atomicAdd(&stats[DM + f + 256], q1);
}

__global__ void bn_finalize_kernel(const float* __restrict__ stats,
                                   const float* __restrict__ gamma,
                                   const float* __restrict__ beta,
                                   float* __restrict__ ab) {
  int f = blockIdx.x * blockDim.x + threadIdx.x;
  if (f < DM) {
    const float invN = 1.f / (float)NTOK;
    float mean = stats[f] * invN;
    float var  = stats[DM + f] * invN - mean * mean;
    float a = gamma[f] * rsqrtf(var + BN_EPS);
    ab[f] = a;
    ab[DM + f] = beta[f] - mean * a;
  }
}

__global__ void bn_apply_kernel(const float* __restrict__ z,
                                const float* __restrict__ ab,
                                float* __restrict__ hO, h16* __restrict__ hH) {
  long i = ((long)blockIdx.x * blockDim.x + threadIdx.x) * 4;
  int f0 = (int)(i & (DM - 1));
  v4f v = *(const v4f*)(z + i);
#pragma unroll
  for (int j = 0; j < 4; ++j) {
    float y = v[j] * ab[f0 + j] + ab[DM + f0 + j];
    hO[i + j] = y;
    hH[i + j] = (h16)y;
  }
}

// ---------------------------------------------------------------------------
// Embedding + positional encoding; weight f32->f16 conversion
// ---------------------------------------------------------------------------
__global__ void embed_kernel(const int* __restrict__ seq,
                             const float* __restrict__ pos,
                             const float* __restrict__ emb,
                             float* __restrict__ h, h16* __restrict__ hh) {
  long i = (long)blockIdx.x * blockDim.x + threadIdx.x;   // over N*D
  long t = i >> 9;
  int d = (int)(i & (DM - 1));
  int s = (int)(t & (SEQ - 1));
  float v = emb[(long)seq[t] * DM + d] + pos[(long)s * DM + d];
  h[i] = v;
  hh[i] = (h16)v;
}

__global__ void f32tof16_kernel(const float* __restrict__ src,
                                h16* __restrict__ dst, int n) {
  int i = blockIdx.x * blockDim.x + threadIdx.x;
  if (i < n) dst[i] = (h16)src[i];
}

// ---------------------------------------------------------------------------
// Host-side orchestration
// ---------------------------------------------------------------------------
extern "C" void kernel_launch(void* const* d_in, const int* in_sizes, int n_in,
                              void* d_out, int out_size, void* d_ws, size_t ws_size,
                              hipStream_t stream) {
  (void)in_sizes; (void)n_in; (void)out_size; (void)ws_size;

  const int*   seq = (const int*)  d_in[0];
  const float* pos = (const float*)d_in[1];
  const float* emb = (const float*)d_in[2];
  const float* Wq  = (const float*)d_in[3];
  const float* bq  = (const float*)d_in[4];
  const float* Wk  = (const float*)d_in[5];
  const float* bk  = (const float*)d_in[6];
  const float* Wv  = (const float*)d_in[7];
  const float* bv  = (const float*)d_in[8];
  const float* Wo  = (const float*)d_in[9];
  const float* bo  = (const float*)d_in[10];
  const float* g1  = (const float*)d_in[11];
  const float* be1 = (const float*)d_in[12];
  const float* W1  = (const float*)d_in[13];
  const float* b1  = (const float*)d_in[14];
  const float* W2  = (const float*)d_in[15];
  const float* b2  = (const float*)d_in[16];
  const float* g2  = (const float*)d_in[17];
  const float* be2 = (const float*)d_in[18];

  char* ws = (char*)d_ws;
  const size_t MB = 1u << 20;
  float* h   = (float*)(ws);                         //  64 MB f32 activations
  float* z   = (float*)(ws + 64  * MB);              //  64 MB f32 pre-BN
  h16*   hh  = (h16*)  (ws + 128 * MB);              //  32 MB f16 activations
  h16*   Qb  = (h16*)  (ws + 160 * MB);              //  32 MB (Q, then attn-out)
  h16*   Kb  = (h16*)  (ws + 192 * MB);              //  32 MB (K, then FFN mid)
  h16*   Vb  = (h16*)  (ws + 224 * MB);              //  32 MB
  h16*   Wq16 = (h16*) (ws + 256 * MB);
  const int WD = DM * DM;                            // 262144 (all weights same)
  h16* Wk16 = Wq16 + WD;
  h16* Wv16 = Wk16 + WD;
  h16* Wo16 = Wv16 + WD;
  h16* W116 = Wo16 + WD;
  h16* W216 = W116 + WD;
  float* stats = (float*)(W216 + WD);                // 1024 f32
  float* ab    = stats + 2 * DM;                     // 1024 f32

  const size_t attn_lds = (size_t)ATTN_LDS_HALVES * sizeof(h16);  // 138240 B
  const dim3 gg(NTOK / 128, DM / 128);               // 256 x 4 GEMM grid

  // embedding + positional
  embed_kernel<<<(NTOK * (long)DM) / 256, 256, 0, stream>>>(seq, pos, emb, h, hh);

  for (int l = 0; l < NLAYER; ++l) {
    // weights -> f16
    f32tof16_kernel<<<WD / 256, 256, 0, stream>>>(Wq + (long)l * WD, Wq16, WD);
    f32tof16_kernel<<<WD / 256, 256, 0, stream>>>(Wk + (long)l * WD, Wk16, WD);
    f32tof16_kernel<<<WD / 256, 256, 0, stream>>>(Wv + (long)l * WD, Wv16, WD);
    f32tof16_kernel<<<WD / 256, 256, 0, stream>>>(Wo + (long)l * WD, Wo16, WD);
    f32tof16_kernel<<<WD / 256, 256, 0, stream>>>(W1 + (long)l * WD, W116, WD);
    f32tof16_kernel<<<WD / 256, 256, 0, stream>>>(W2 + (long)l * WD, W216, WD);

    // Q/K/V projections
    gemm_kernel<<<gg, 256, 0, stream>>>(hh, Wq16, bq + l * DM, nullptr, nullptr, Qb, 0);
    gemm_kernel<<<gg, 256, 0, stream>>>(hh, Wk16, bk + l * DM, nullptr, nullptr, Kb, 0);
    gemm_kernel<<<gg, 256, 0, stream>>>(hh, Wv16, bv + l * DM, nullptr, nullptr, Vb, 0);

    // fused attention (writes back into Qb)
    attn_kernel<<<dim3(128 * NHEAD, 2), 256, attn_lds, stream>>>(Qb, Kb, Vb, Qb);

    // output projection + residual -> z (f32)
    gemm_kernel<<<gg, 256, 0, stream>>>(Qb, Wo16, bo + l * DM, h, z, nullptr, 2);

    // BatchNorm 1
    (void)hipMemsetAsync(stats, 0, 2 * DM * sizeof(float), stream);
    bn_stats_kernel<<<NTOK / 128, 256, 0, stream>>>(z, stats);
    bn_finalize_kernel<<<2, 256, 0, stream>>>(stats, g1 + l * DM, be1 + l * DM, ab);
    bn_apply_kernel<<<(NTOK * (long)DM) / 1024, 256, 0, stream>>>(z, ab, h, hh);

    // FFN
    gemm_kernel<<<gg, 256, 0, stream>>>(hh, W116, b1 + l * DM, nullptr, nullptr, Kb, 1);
    gemm_kernel<<<gg, 256, 0, stream>>>(Kb, W216, b2 + l * DM, h, z, nullptr, 2);

    // BatchNorm 2 (final layer writes f32 result straight to d_out)
    (void)hipMemsetAsync(stats, 0, 2 * DM * sizeof(float), stream);
    bn_stats_kernel<<<NTOK / 128, 256, 0, stream>>>(z, stats);
    bn_finalize_kernel<<<2, 256, 0, stream>>>(stats, g2 + l * DM, be2 + l * DM, ab);
    float* dst = (l == NLAYER - 1) ? (float*)d_out : h;
    bn_apply_kernel<<<(NTOK * (long)DM) / 1024, 256, 0, stream>>>(z, ab, dst, hh);
  }
}
